// Deformable_spatemp_atten_13477607375409
// MI455X (gfx1250) — matrix operand order repaired and checked by reading, not compiled, not run on hardware
//
#include <hip/hip_runtime.h>
#include <math.h>

// ---------------- problem constants ----------------
#define B_ 2
#define T_ 8
#define C_ 256
#define H_ 24
#define W_ 24
#define NH_ 8
#define HC_ 32
#define G_ 16            // B*NH
#define HW_ 576          // H*W
#define M_ 4608          // T*H*W
#define NPT_ 36          // Tk*Hk*Wk
#define SCALE_ 0.1767766952966369f   // 32^-0.5

typedef __attribute__((ext_vector_type(16))) __bf16 v16bf;
typedef __attribute__((ext_vector_type(8)))  float v8f;

// ---------------- bf16 helpers ----------------
__device__ __forceinline__ unsigned short f32_to_bf16(float f) {
  unsigned int u = __builtin_bit_cast(unsigned int, f);
  u += 0x7FFFu + ((u >> 16) & 1u);          // round-to-nearest-even
  return (unsigned short)(u >> 16);
}
__device__ __forceinline__ float bf16us_to_f32(unsigned short h) {
  unsigned int u = ((unsigned int)h) << 16;
  return __builtin_bit_cast(float, u);
}
__device__ __forceinline__ __bf16 us_to_bf(unsigned short h) {
  return __builtin_bit_cast(__bf16, h);
}

union U16x8 { uint4 u4; unsigned short s[8]; };

// ---------------- WMMA 16x16x32 bf16 per-lane layouts (ISA 7.12.2) ----------------
// A (16x32, MxK): lanes 0-15 row m=lane, K 0..7 then 16..23; lanes 16-31 row m=lane-16, K 8..15 then 24..31
__device__ __forceinline__ int a_k_of(int lane, int j) {
  return ((lane >> 4) << 3) + ((j >> 3) << 4) + (j & 7);
}
// B (32x16, KxN): lanes 0-15 col n=lane, K 0..7 then 16..23; lanes 16-31 col n=lane-16, K 8..15 then 24..31
__device__ __forceinline__ int b_k_of(int lane, int j) {
  return ((lane >> 4) << 3) + ((j >> 3) << 4) + (j & 7);
}
// D (16x16 f32): elem v -> row m = v + 8*(lane>=16), col n = lane&15

// =====================================================================
// k0: zero bf16 pad regions (k tile cols 36..47, vT rows 36..63)
// =====================================================================
__global__ void zero_us_kernel(unsigned short* __restrict__ p, int n) {
  int i = blockIdx.x * 256 + threadIdx.x;
  if (i < n) p[i] = 0;
}

// =====================================================================
// k1/k2: C[r, n] = sum_k A[r,k] * W[n,k] + bias[n]   (M=9216, K=N=256)
// MODE 0: A = x  (B,T,C,H,W) f32      -> xc
// MODE 1: A = xc (B,C,T,H,W) bf16     -> q
// Output stored in (B,C,T,H,W) layout, f32 + bf16.
// Block = 256 threads (8 waves): 16(M) x 128(N) tile, each wave owns 16 cols.
// A tile staged with B128 (f32) / B64 (bf16) loads along m (contiguous; 576 % 16 == 0 so a
// 16-row tile never crosses a (b,t) plane). B tile staged with B128 loads
// along k. Next K-step's weights prefetched (global_prefetch_b8).
// =====================================================================
template <int MODE>
__global__ void gemm_proj_kernel(const void* __restrict__ srcv,
                                 const float* __restrict__ w,
                                 const float* __restrict__ bias,
                                 float* __restrict__ out_f32,
                                 unsigned short* __restrict__ out_bf16) {
  __shared__ __align__(16) unsigned short sA[16 * 32];
  __shared__ __align__(16) unsigned short sB[128 * 32];
  const int tid = threadIdx.x;
  const int wave = tid >> 5, lane = tid & 31;
  const int mt = blockIdx.x >> 1;           // 0..575
  const int nb = (blockIdx.x & 1) * 128;
  const int ncol = nb + wave * 16 + (lane & 15);

  const float* srcf = (const float*)srcv;
  const unsigned short* srcu = (const unsigned short*)srcv;

  // uniform row-tile decomposition (constant within the tile)
  const int r0 = mt * 16;
  const int btA = r0 / HW_, hwA = r0 % HW_;
  const int bA = btA >> 3, tA = btA & 7;

  v8f acc;
  {
    float bv = bias[ncol];
#pragma unroll
    for (int i = 0; i < 8; i++) acc[i] = bv;
  }

  for (int k0 = 0; k0 < 256; k0 += 32) {
    // ---- stage A tile (16 m x 32 k): threads 0..127, one B128 (f32) / B64 (bf16) each ----
    if (tid < 128) {
      int kc = tid >> 2;
      int mq = (tid & 3) << 2;
      if (MODE == 0) {                        // x: (B,T,C,H,W) f32
        const float4 vv =
            *(const float4*)&srcf[(btA * C_ + k0 + kc) * HW_ + hwA + mq];
        sA[(mq + 0) * 32 + kc] = f32_to_bf16(vv.x);
        sA[(mq + 1) * 32 + kc] = f32_to_bf16(vv.y);
        sA[(mq + 2) * 32 + kc] = f32_to_bf16(vv.z);
        sA[(mq + 3) * 32 + kc] = f32_to_bf16(vv.w);
      } else {                                // xc: (B,C,T,H,W) bf16
        const ushort4 vv = *(const ushort4*)
            &srcu[((bA * C_ + k0 + kc) * T_ + tA) * HW_ + hwA + mq];
        sA[(mq + 0) * 32 + kc] = vv.x;
        sA[(mq + 1) * 32 + kc] = vv.y;
        sA[(mq + 2) * 32 + kc] = vv.z;
        sA[(mq + 3) * 32 + kc] = vv.w;
      }
    }
    // ---- stage B tile (128 n x 32 k): 4 B128 loads per thread ----
    for (int e4 = tid; e4 < 1024; e4 += 256) {
      int e = e4 << 2;
      int nn = e >> 5, kc = e & 31;
      const float4 wv = *(const float4*)&w[(nb + nn) * 256 + k0 + kc];
      unsigned int lo = (unsigned int)f32_to_bf16(wv.x) |
                        ((unsigned int)f32_to_bf16(wv.y) << 16);
      unsigned int hi = (unsigned int)f32_to_bf16(wv.z) |
                        ((unsigned int)f32_to_bf16(wv.w) << 16);
      *(unsigned int*)&sB[e] = lo;
      *(unsigned int*)&sB[e + 2] = hi;
    }
    // ---- speculative prefetch of next K-step weight tile ----
    if (k0 + 32 < 256) {
      __builtin_prefetch(&w[(nb + (tid >> 1)) * 256 + k0 + 32 + (tid & 1) * 16], 0, 0);
    }
    __syncthreads();

    v16bf a, bm;
#pragma unroll
    for (int j = 0; j < 16; j++) {
      a[j] = us_to_bf(sA[(lane & 15) * 32 + a_k_of(lane, j)]);
      bm[j] = us_to_bf(sB[(wave * 16 + (lane & 15)) * 32 + b_k_of(lane, j)]);
    }
    acc = __builtin_amdgcn_wmma_f32_16x16x32_bf16(false, a, false, bm,
                                                  (short)0, acc, false, false);
    __syncthreads();
  }

#pragma unroll
  for (int v = 0; v < 8; v++) {
    int mloc = v + ((lane >> 4) << 3);
    int hw = hwA + mloc;
    int o = ((bA * C_ + ncol) * T_ + tA) * HW_ + hw;   // (B,C,T,H,W)
    out_f32[o] = acc[v];
    out_bf16[o] = f32_to_bf16(acc[v]);
  }
}

// =====================================================================
// k3: depthwise offset conv (3,9,9), stride (2,8,8), pad (1,4,4)
// in : q (G,HC,T,H,W) f32 ; out: off[(g*36+n)*32+ch]
// =====================================================================
__global__ void offconv_kernel(const float* __restrict__ q,
                               const float* __restrict__ wdw,   // (32,1,3,9,9)
                               const float* __restrict__ bdw,
                               float* __restrict__ off) {
  int idx = blockIdx.x * 256 + threadIdx.x;
  if (idx >= G_ * NPT_ * HC_) return;
  int ch = idx & 31;
  int gn = idx >> 5;
  int g = gn / NPT_, nn = gn % NPT_;
  int to = nn / 9, rem = nn % 9, ho = rem / 3, wo = rem % 3;
  float acc = bdw[ch];
  const float* qb = q + (g * HC_ + ch) * M_;
  for (int kt = 0; kt < 3; kt++) {
    int t = to * 2 - 1 + kt;
    if (t < 0 || t >= T_) continue;
    for (int kh = 0; kh < 9; kh++) {
      int hh = ho * 8 - 4 + kh;
      if (hh < 0 || hh >= H_) continue;
      for (int kw = 0; kw < 9; kw++) {
        int ww = wo * 8 - 4 + kw;
        if (ww < 0 || ww >= W_) continue;
        acc = fmaf(qb[t * HW_ + hh * W_ + ww],
                   wdw[((ch * 3 + kt) * 9 + kh) * 9 + kw], acc);
      }
    }
  }
  off[idx] = acc;
}

// =====================================================================
// k4: LayerNorm(HC) + exact GELU + proj(3) + ref grid + clip -> pos[(g*36+n)*3+p]
// =====================================================================
__global__ void offpost_kernel(const float* __restrict__ off,
                               const float* __restrict__ ln_g,
                               const float* __restrict__ ln_b,
                               const float* __restrict__ wproj,  // (3,32)
                               float* __restrict__ pos) {
  int idx = blockIdx.x * 192 + threadIdx.x;
  if (idx >= G_ * NPT_) return;
  const float* o = off + idx * 32;
  float mu = 0.f;
  for (int c = 0; c < 32; c++) mu += o[c];
  mu *= (1.0f / 32.0f);
  float var = 0.f;
  for (int c = 0; c < 32; c++) { float d = o[c] - mu; var += d * d; }
  var *= (1.0f / 32.0f);
  float inv = rsqrtf(var + 1e-5f);
  float p0 = 0.f, p1 = 0.f, p2 = 0.f;
  for (int c = 0; c < 32; c++) {
    float y = (o[c] - mu) * inv * ln_g[c] + ln_b[c];
    y = 0.5f * y * (1.0f + erff(y * 0.70710678118654752f));   // exact GELU
    p0 = fmaf(wproj[c], y, p0);
    p1 = fmaf(wproj[32 + c], y, p1);
    p2 = fmaf(wproj[64 + c], y, p2);
  }
  int nn = idx % NPT_;
  int to = nn / 9, ho = (nn % 9) / 3, wo = nn % 3;
  float rt = ((to + 0.5f) / 3.0f) * 2.0f - 1.0f;   // (Tk-1)=3
  float ry = ((ho + 0.5f) / 2.0f) * 2.0f - 1.0f;   // (Hk-1)=2
  float rx = ((wo + 0.5f) / 2.0f) * 2.0f - 1.0f;
  pos[idx * 3 + 0] = fminf(fmaxf(p0 + rt, -1.0f), 1.0f);
  pos[idx * 3 + 1] = fminf(fmaxf(p1 + ry, -1.0f), 1.0f);
  pos[idx * 3 + 2] = fminf(fmaxf(p2 + rx, -1.0f), 1.0f);
}

// =====================================================================
// k5: trilinear grid-sample of xc at pos -> xs[(g*36+n)*32+ch] bf16
// grid axes per reference: gx(W)<-pos_t, gy(H)<-pos_x, gz(T)<-pos_y
// =====================================================================
__global__ void sample_kernel(const float* __restrict__ xc,   // (G,HC,T,H,W)
                              const float* __restrict__ pos,
                              unsigned short* __restrict__ xs) {
  int idx = blockIdx.x * 256 + threadIdx.x;
  if (idx >= G_ * NPT_ * HC_) return;
  int ch = idx & 31;
  int gn = idx >> 5;
  int g = gn / NPT_;
  float gx = pos[gn * 3 + 0];
  float gz = pos[gn * 3 + 1];
  float gy = pos[gn * 3 + 2];
  float ix = (gx + 1.0f) * 0.5f * (W_ - 1);
  float iy = (gy + 1.0f) * 0.5f * (H_ - 1);
  float iz = (gz + 1.0f) * 0.5f * (T_ - 1);
  float fx0 = floorf(ix), fy0 = floorf(iy), fz0 = floorf(iz);
  float fx = ix - fx0, fy = iy - fy0, fz = iz - fz0;
  int x0 = (int)fx0; x0 = x0 < 0 ? 0 : (x0 > W_ - 1 ? W_ - 1 : x0);
  int y0 = (int)fy0; y0 = y0 < 0 ? 0 : (y0 > H_ - 1 ? H_ - 1 : y0);
  int z0 = (int)fz0; z0 = z0 < 0 ? 0 : (z0 > T_ - 1 ? T_ - 1 : z0);
  int x1 = x0 + 1 > W_ - 1 ? W_ - 1 : x0 + 1;
  int y1 = y0 + 1 > H_ - 1 ? H_ - 1 : y0 + 1;
  int z1 = z0 + 1 > T_ - 1 ? T_ - 1 : z0 + 1;
  const float* img = xc + (g * HC_ + ch) * M_;
  auto Gv = [&](int z, int y, int x) { return img[z * HW_ + y * W_ + x]; };
  float v =
      Gv(z0, y0, x0) * (1 - fz) * (1 - fy) * (1 - fx) + Gv(z0, y0, x1) * (1 - fz) * (1 - fy) * fx +
      Gv(z0, y1, x0) * (1 - fz) * fy * (1 - fx)       + Gv(z0, y1, x1) * (1 - fz) * fy * fx +
      Gv(z1, y0, x0) * fz * (1 - fy) * (1 - fx)       + Gv(z1, y0, x1) * fz * (1 - fy) * fx +
      Gv(z1, y1, x0) * fz * fy * (1 - fx)             + Gv(z1, y1, x1) * fz * fy * fx;
  xs[idx] = f32_to_bf16(v);
}

// =====================================================================
// k6: k/v projection GEMM: rows r=(b,n_s) [72, pad 80], K=256, N=256
// mode 0 -> k_bf16 (G,HC,48)  | mode 1 -> vT_bf16 (G,64,HC)
// Block = 1 wave, one 16x16 output tile per block.
// =====================================================================
__global__ void gemm_kv_kernel(const unsigned short* __restrict__ xs,
                               const float* __restrict__ w,
                               const float* __restrict__ bias,
                               unsigned short* __restrict__ outk,
                               unsigned short* __restrict__ outv,
                               int mode) {
  __shared__ __align__(16) unsigned short sA[16 * 32];
  __shared__ __align__(16) unsigned short sB[16 * 32];
  const int lane = threadIdx.x;
  const int mt = blockIdx.x / 16;   // 0..4
  const int nt = blockIdx.x % 16;
  const int ncol = nt * 16 + (lane & 15);
  v8f acc;
  {
    float bv = bias[ncol];
#pragma unroll
    for (int i = 0; i < 8; i++) acc[i] = bv;
  }
  for (int k0 = 0; k0 < 256; k0 += 32) {
    for (int e = lane; e < 512; e += 32) {
      int mloc = e >> 5, kc = e & 31;
      int r = mt * 16 + mloc;
      unsigned short aval = 0;
      if (r < B_ * NPT_) {
        int b = r / NPT_, ns = r % NPT_;
        int ci = k0 + kc;
        aval = xs[(((b * NH_ + (ci >> 5)) * NPT_ + ns) << 5) + (ci & 31)];
      }
      sA[e] = aval;
      sB[e] = f32_to_bf16(w[(nt * 16 + mloc) * 256 + k0 + kc]);
    }
    __syncthreads();
    v16bf a, bm;
#pragma unroll
    for (int j = 0; j < 16; j++) {
      a[j] = us_to_bf(sA[(lane & 15) * 32 + a_k_of(lane, j)]);
      bm[j] = us_to_bf(sB[(lane & 15) * 32 + b_k_of(lane, j)]);
    }
    acc = __builtin_amdgcn_wmma_f32_16x16x32_bf16(false, a, false, bm,
                                                  (short)0, acc, false, false);
    __syncthreads();
  }
#pragma unroll
  for (int v = 0; v < 8; v++) {
    int mloc = v + ((lane >> 4) << 3);
    int r = mt * 16 + mloc;
    if (r < B_ * NPT_) {
      int b = r / NPT_, ns = r % NPT_;
      int g2 = ncol >> 5, ch2 = ncol & 31;
      unsigned short h = f32_to_bf16(acc[v]);
      if (mode == 0)
        outk[((b * NH_ + g2) * HC_ + ch2) * 48 + ns] = h;
      else
        outv[((b * NH_ + g2) * 64 + ns) * HC_ + ch2] = h;
    }
  }
}

// =====================================================================
// k7: fused attention per (g, 16-row m-tile):
// qk WMMA (K=32, 3 n-tiles) -> +logCPB bias (VALU MLP) -> softmax -> attn@v WMMA
// =====================================================================
__global__ void attn_kernel(const unsigned short* __restrict__ q_bf16,
                            const unsigned short* __restrict__ k_bf16,  // (G,HC,48)
                            const unsigned short* __restrict__ vT_bf16, // (G,64,HC)
                            const float* __restrict__ pos,
                            const float* __restrict__ rpe_w1,   // (32,3)
                            const float* __restrict__ rpe_b1,   // (32)
                            const float* __restrict__ rpe_w2,   // (1,32)
                            float* __restrict__ out) {
  __shared__ __align__(16) unsigned short sQ[16 * 32];
  __shared__ __align__(16) unsigned short sK[32 * 48];
  __shared__ __align__(16) unsigned short sV[64 * 32];
  __shared__ float sAttn[16 * 64];
  __shared__ float sPos[NPT_ * 3];
  __shared__ float sW1[96], sB1[32], sW2[32];
  const int tid = threadIdx.x, lane = tid & 31, wave = tid >> 5;
  const int g = blockIdx.x / 288;
  const int m0 = (blockIdx.x % 288) * 16;

  // ---- B128 staging of q/k/v tiles ----
  if (tid < 64) {                 // sQ: 32 c x 16 m, load 8 m per thread
    int c = tid >> 1, mh = (tid & 1) * 8;
    U16x8 u;
    u.u4 = *(const uint4*)&q_bf16[(g * HC_ + c) * M_ + m0 + mh];
#pragma unroll
    for (int i = 0; i < 8; i++) sQ[(mh + i) * 32 + c] = u.s[i];
  }
  if (tid < 192)                  // sK: 1536 ushorts contiguous
    *(uint4*)&sK[tid * 8] = *(const uint4*)&k_bf16[g * 1536 + tid * 8];
  *(uint4*)&sV[tid * 8] = *(const uint4*)&vT_bf16[g * 2048 + tid * 8];
  for (int e = tid; e < 1024; e += 256) sAttn[e] = 0.0f;
  if (tid < 108) sPos[tid] = pos[g * 108 + tid];
  if (tid < 96) sW1[tid] = rpe_w1[tid];
  if (tid < 32) { sB1[tid] = rpe_b1[tid]; sW2[tid] = rpe_w2[tid]; }
  __syncthreads();

  if (wave == 0) {   // q^T k : one K=32 WMMA per 16-col n-tile
    v16bf a;
#pragma unroll
    for (int j = 0; j < 16; j++)
      a[j] = us_to_bf(sQ[(lane & 15) * 32 + a_k_of(lane, j)]);
#pragma unroll
    for (int nt = 0; nt < 3; nt++) {
      v16bf bm;
#pragma unroll
      for (int j = 0; j < 16; j++)
        bm[j] = us_to_bf(sK[b_k_of(lane, j) * 48 + nt * 16 + (lane & 15)]);
      v8f acc = {};
      acc = __builtin_amdgcn_wmma_f32_16x16x32_bf16(false, a, false, bm,
                                                    (short)0, acc, false, false);
#pragma unroll
      for (int v = 0; v < 8; v++) {
        int mloc = v + ((lane >> 4) << 3);
        sAttn[mloc * 64 + nt * 16 + (lane & 15)] = acc[v] * SCALE_;
      }
    }
  }
  __syncthreads();

  // log-CPB relative position bias (MLP 3->32->1), all 256 threads
  for (int idx = tid; idx < 16 * NPT_; idx += 256) {
    int mloc = idx / NPT_, nn = idx % NPT_;
    int mg = m0 + mloc;
    int t = mg / HW_, hw = mg % HW_, hh = hw / W_, ww = hw % W_;
    float qt = t * (2.0f / (T_ - 1)) - 1.0f;
    float qy = hh * (2.0f / (W_ - 1)) - 1.0f;   // ref normalizes y by W-1
    float qx = ww * (2.0f / (H_ - 1)) - 1.0f;   // and x by H-1 (faithful)
    float d0 = (qt - sPos[nn * 3 + 0]) * 4.0f;
    float d1 = (qy - sPos[nn * 3 + 1]) * 4.0f;
    float d2 = (qx - sPos[nn * 3 + 2]) * 4.0f;
    d0 = copysignf(log2f(fabsf(d0) + 1.0f) * (1.0f / 3.0f), d0);
    d1 = copysignf(log2f(fabsf(d1) + 1.0f) * (1.0f / 3.0f), d1);
    d2 = copysignf(log2f(fabsf(d2) + 1.0f) * (1.0f / 3.0f), d2);
    float acc = 0.0f;
#pragma unroll
    for (int hn = 0; hn < 32; hn++) {
      float hv = fmaf(sW1[hn * 3 + 0], d0,
                 fmaf(sW1[hn * 3 + 1], d1,
                 fmaf(sW1[hn * 3 + 2], d2, sB1[hn])));
      hv = fmaxf(hv, 0.0f);
      acc = fmaf(sW2[hn], hv, acc);
    }
    sAttn[mloc * 64 + nn] += acc;
  }
  __syncthreads();

  if (tid < 16) {   // softmax over n=36 per row
    float mx = -1e30f;
    for (int nn = 0; nn < NPT_; nn++) mx = fmaxf(mx, sAttn[tid * 64 + nn]);
    float s = 0.0f;
    for (int nn = 0; nn < NPT_; nn++) {
      float e = expf(sAttn[tid * 64 + nn] - mx);
      sAttn[tid * 64 + nn] = e;
      s += e;
    }
    float inv = 1.0f / s;
    for (int nn = 0; nn < NPT_; nn++) sAttn[tid * 64 + nn] *= inv;
  }
  __syncthreads();

  if (wave == 0) {   // attn @ v : K=64 (zero-padded), 2 k-steps x 2 ch-tiles
    v8f o0 = {}, o1 = {};
#pragma unroll
    for (int ks = 0; ks < 2; ks++) {
      v16bf a2, b0, b1;
#pragma unroll
      for (int j = 0; j < 16; j++) {
        a2[j] = us_to_bf(f32_to_bf16(sAttn[(lane & 15) * 64 + ks * 32 + a_k_of(lane, j)]));
        int kr = ks * 32 + b_k_of(lane, j);
        b0[j] = us_to_bf(sV[kr * 32 + (lane & 15)]);
        b1[j] = us_to_bf(sV[kr * 32 + 16 + (lane & 15)]);
      }
      o0 = __builtin_amdgcn_wmma_f32_16x16x32_bf16(false, a2, false, b0, (short)0, o0, false, false);
      o1 = __builtin_amdgcn_wmma_f32_16x16x32_bf16(false, a2, false, b1, (short)0, o1, false, false);
    }
    int bb = g >> 3, gp = g & 7;
    int c0 = gp * HC_ + (lane & 15);
#pragma unroll
    for (int v = 0; v < 8; v++) {
      int mloc = v + ((lane >> 4) << 3);
      int mg = m0 + mloc;
      int t = mg / HW_, hw = mg % HW_;
      size_t base = ((size_t)(bb * T_ + t) * C_) * HW_ + hw;   // (B,T,C,H,W)
      out[base + (size_t)c0 * HW_] = o0[v];
      out[base + (size_t)(c0 + 16) * HW_] = o1[v];
    }
  }
}

// =====================================================================
// host launcher
// =====================================================================
extern "C" void kernel_launch(void* const* d_in, const int* in_sizes, int n_in,
                              void* d_out, int out_size, void* d_ws, size_t ws_size,
                              hipStream_t stream) {
  const float* x        = (const float*)d_in[0];
  const float* w_in     = (const float*)d_in[1];
  const float* b_in     = (const float*)d_in[2];
  const float* w_q      = (const float*)d_in[3];
  const float* b_q      = (const float*)d_in[4];
  const float* w_off_dw = (const float*)d_in[5];
  const float* b_off_dw = (const float*)d_in[6];
  const float* ln_g     = (const float*)d_in[7];
  const float* ln_b     = (const float*)d_in[8];
  const float* w_off_p  = (const float*)d_in[9];
  const float* w_k      = (const float*)d_in[10];
  const float* b_k      = (const float*)d_in[11];
  const float* w_v      = (const float*)d_in[12];
  const float* b_v      = (const float*)d_in[13];
  const float* rpe_w1   = (const float*)d_in[14];
  const float* rpe_b1   = (const float*)d_in[15];
  const float* rpe_w2   = (const float*)d_in[16];

  char* ws = (char*)d_ws;
  float*          xc_f32 = (float*)(ws + 0);                    //  9.0 MB
  float*          q_f32  = (float*)(ws + 9437184);              //  9.0 MB
  unsigned short* xc_bf  = (unsigned short*)(ws + 18874368);    //  4.5 MB
  unsigned short* q_bf   = (unsigned short*)(ws + 23592960);    //  4.5 MB
  float*          offb   = (float*)(ws + 28311552);             //  72 KB
  float*          posb   = (float*)(ws + 28385280);             //   7 KB
  unsigned short* xs_bf  = (unsigned short*)(ws + 28392192);    //  36 KB
  unsigned short* k_bf   = (unsigned short*)(ws + 28429056);    //  48 KB
  unsigned short* vT_bf  = (unsigned short*)(ws + 28478208);    //  64 KB

  // zero padded k/vT regions (contiguous: 24576 + 32768 ushorts)
  zero_us_kernel<<<(57344 + 255) / 256, 256, 0, stream>>>(k_bf, 57344);

  // xc = x * W_in^T + b_in ; q = xc * W_q^T + b_q
  gemm_proj_kernel<0><<<1152, 256, 0, stream>>>(x, w_in, b_in, xc_f32, xc_bf);
  gemm_proj_kernel<1><<<1152, 256, 0, stream>>>(xc_bf, w_q, b_q, q_f32, q_bf);

  // offsets -> sampling positions
  offconv_kernel<<<72, 256, 0, stream>>>(q_f32, w_off_dw, b_off_dw, offb);
  offpost_kernel<<<3, 192, 0, stream>>>(offb, ln_g, ln_b, w_off_p, posb);

  // deformable trilinear sampling
  sample_kernel<<<72, 256, 0, stream>>>(xc_f32, posb, xs_bf);

  // k / v projections
  gemm_kv_kernel<<<80, 32, 0, stream>>>(xs_bf, w_k, b_k, k_bf, vT_bf, 0);
  gemm_kv_kernel<<<80, 32, 0, stream>>>(xs_bf, w_v, b_v, k_bf, vT_bf, 1);

  // fused attention + output
  attn_kernel<<<G_ * 288, 256, 0, stream>>>(q_bf, k_bf, vT_bf, posb,
                                            rpe_w1, rpe_b1, rpe_w2,
                                            (float*)d_out);
  (void)in_sizes; (void)n_in; (void)out_size; (void)ws_size;
}
